// BitTransformerLayer_68118181314704
// MI455X (gfx1250) — 16-bit and 8-bit matrix operand order repaired and checked by reading, compiled, not run on hardware
//
#include <hip/hip_runtime.h>
#include <hip/hip_bf16.h>


typedef __attribute__((ext_vector_type(16))) _Float16 v16h;
typedef __attribute__((ext_vector_type(8)))  _Float16 v8h;
typedef __attribute__((ext_vector_type(8)))  float    v8f;
typedef __attribute__((ext_vector_type(8)))  int      v8i;

#define BB 8
#define SS 1024
#define DD 1024
#define HH 16
#define DH 64
#define FFN 4096
#define MM (BB*SS)
#define EPSF 1e-6f

union F16Frag { v16h v; struct { v8h lo, hi; } s; };

template <int CTRL>
__device__ __forceinline__ float dpp_xor(float x) {
  return __int_as_float(__builtin_amdgcn_update_dpp(
      0, __float_as_int(x), CTRL, 0xF, 0xF, true));
}
__device__ __forceinline__ float row16_max(float x) {
  x = fmaxf(x, dpp_xor<0xB1>(x));
  x = fmaxf(x, dpp_xor<0x4E>(x));
  x = fmaxf(x, dpp_xor<0x141>(x));
  x = fmaxf(x, dpp_xor<0x140>(x));
  return x;
}
__device__ __forceinline__ float row16_sum(float x) {
  x += dpp_xor<0xB1>(x);
  x += dpp_xor<0x4E>(x);
  x += dpp_xor<0x141>(x);
  x += dpp_xor<0x140>(x);
  return x;
}

__global__ void k_f32_to_f16(const float* __restrict__ in, _Float16* __restrict__ out, int n) {
  for (int i = blockIdx.x * blockDim.x + threadIdx.x; i < n; i += gridDim.x * blockDim.x)
    out[i] = (_Float16)in[i];
}

__global__ void k_abs_sum(const float* __restrict__ w, int n, float* __restrict__ accum) {
  float s = 0.f;
  for (int i = blockIdx.x * blockDim.x + threadIdx.x; i < n; i += gridDim.x * blockDim.x)
    s += fabsf(w[i]);
  __shared__ float red[256];
  int t = threadIdx.x;
  red[t] = s; __syncthreads();
  for (int off = 128; off > 0; off >>= 1) { if (t < off) red[t] += red[t + off]; __syncthreads(); }
  if (t == 0) atomicAdd(accum, red[0]);
}

__global__ void k_tern_quant(const float* __restrict__ w, int n,
                             const float* __restrict__ accum, float invCount,
                             signed char* __restrict__ q, float* __restrict__ wscale) {
  float mean = fmaxf(accum[0] * invCount, 1e-5f);
  float s = 1.f / mean;
  int gid = blockIdx.x * blockDim.x + threadIdx.x;
  for (int i = gid; i < n; i += gridDim.x * blockDim.x) {
    float r = roundf(w[i] * s);
    r = fminf(1.f, fmaxf(-1.f, r));
    q[i] = (signed char)r;
  }
  if (gid == 0) wscale[0] = mean;
}

__global__ void k_rmsnorm_f16(const float* __restrict__ x, const float* __restrict__ w,
                              _Float16* __restrict__ out) {
  int row = blockIdx.x, t = threadIdx.x;
  const float* xr = x + (size_t)row * DD;
  float v[4]; float ss = 0.f;
  for (int i = 0; i < 4; i++) { v[i] = xr[t + 256 * i]; ss += v[i] * v[i]; }
  __shared__ float red[256];
  red[t] = ss; __syncthreads();
  for (int off = 128; off > 0; off >>= 1) { if (t < off) red[t] += red[t + off]; __syncthreads(); }
  float inv = rsqrtf(red[0] * (1.f / DD) + EPSF);
  _Float16* orow = out + (size_t)row * DD;
  for (int i = 0; i < 4; i++) { int c = t + 256 * i; orow[c] = (_Float16)(v[i] * inv * w[c]); }
}

__global__ void k_rmsnorm_q8(const float* __restrict__ x, const float* __restrict__ w,
                             signed char* __restrict__ q, float* __restrict__ rowscale) {
  int row = blockIdx.x, t = threadIdx.x;
  const float* xr = x + (size_t)row * DD;
  float v[4]; float ss = 0.f;
  for (int i = 0; i < 4; i++) { v[i] = xr[t + 256 * i]; ss += v[i] * v[i]; }
  __shared__ float red[256];
  red[t] = ss; __syncthreads();
  for (int off = 128; off > 0; off >>= 1) { if (t < off) red[t] += red[t + off]; __syncthreads(); }
  float inv = rsqrtf(red[0] * (1.f / DD) + EPSF);
  float y[4]; float amax = 0.f;
  for (int i = 0; i < 4; i++) {
    int c = t + 256 * i;
    y[i] = v[i] * inv * w[c];
    amax = fmaxf(amax, fabsf(y[i]));
  }
  __syncthreads();
  red[t] = amax; __syncthreads();
  for (int off = 128; off > 0; off >>= 1) { if (t < off) red[t] = fmaxf(red[t], red[t + off]); __syncthreads(); }
  float ax = fmaxf(red[0], 1e-5f);
  float s = 127.f / ax;
  signed char* qr = q + (size_t)row * DD;
  for (int i = 0; i < 4; i++) {
    int c = t + 256 * i;
    float r = fminf(127.f, fmaxf(-128.f, roundf(y[i] * s)));
    qr[c] = (signed char)r;
  }
  if (t == 0) rowscale[row] = ax * (1.f / 127.f);
}

__global__ void k_actquant_f16(const _Float16* __restrict__ x, signed char* __restrict__ q,
                               float* __restrict__ rowscale, int cols) {
  int row = blockIdx.x, t = threadIdx.x;
  const _Float16* xr = x + (size_t)row * cols;
  float amax = 0.f;
  for (int c = t; c < cols; c += 256) amax = fmaxf(amax, fabsf((float)xr[c]));
  __shared__ float red[256];
  red[t] = amax; __syncthreads();
  for (int off = 128; off > 0; off >>= 1) { if (t < off) red[t] = fmaxf(red[t], red[t + off]); __syncthreads(); }
  float ax = fmaxf(red[0], 1e-5f);
  float s = 127.f / ax;
  signed char* qr = q + (size_t)row * cols;
  for (int c = t; c < cols; c += 256) {
    float r = fminf(127.f, fmaxf(-128.f, roundf((float)xr[c] * s)));
    qr[c] = (signed char)r;
  }
  if (t == 0) rowscale[row] = ax * (1.f / 127.f);
}

template <int MODE>
__global__ void k_gemm_f16(const _Float16* __restrict__ A, const _Float16* __restrict__ W,
                           const float* __restrict__ bias, const float* __restrict__ resid,
                           _Float16* __restrict__ outh, float* __restrict__ outf,
                           int Ndim, int Kdim) {
  const int lane = threadIdx.x & 31;
  const int wave = threadIdx.x >> 5;
  const int half = lane >> 4;
  const int l16  = lane & 15;
  const int nBase = blockIdx.x * 64;
  const int mBase = blockIdx.y * 128 + wave * 16;

  v8f acc[4] = {};
  const _Float16* arowBase = A + (size_t)(mBase + l16) * Kdim + half * 16;
  for (int k = 0; k < Kdim; k += 32) {
    F16Frag af;
    const _Float16* arow = arowBase + k;
    af.s.lo = *(const v8h*)(arow);
    af.s.hi = *(const v8h*)(arow + 8);
    if (k + 64 < Kdim) __builtin_prefetch(arow + 64, 0, 1);
    for (int t = 0; t < 4; t++) {
      const _Float16* wrow = W + (size_t)(nBase + t * 16 + l16) * Kdim + k + half * 16;
      v16h bf = *(const v16h*)wrow;
      acc[t] = __builtin_amdgcn_wmma_f32_16x16x32_f16(false, af.v, false, bf,
                                                      (short)0, acc[t], false, false);
    }
  }
  for (int t = 0; t < 4; t++) {
    int n = nBase + t * 16 + l16;
    float bn = bias ? bias[n] : 0.f;
    for (int v = 0; v < 8; v++) {
      int m = mBase + v + half * 8;
      float val = acc[t][v] + bn;
      if (MODE == 0) outh[(size_t)m * Ndim + n] = (_Float16)val;
      else           outf[(size_t)m * Ndim + n] = resid[(size_t)m * Ndim + n] + val;
    }
  }
}

template <int MODE>
__global__ void k_gemm_i8(const signed char* __restrict__ A, const signed char* __restrict__ W,
                          const float* __restrict__ rowscale, const float* __restrict__ wscale,
                          const float* __restrict__ bias, const float* __restrict__ resid,
                          _Float16* __restrict__ outh, float* __restrict__ outf,
                          int Ndim, int Kdim) {
  const int lane = threadIdx.x & 31;
  const int wave = threadIdx.x >> 5;
  const int half = lane >> 4;
  const int l16  = lane & 15;
  const int nBase = blockIdx.x * 64;
  const int mBase = blockIdx.y * 128 + wave * 16;

  v8i acc[4] = {};
  const signed char* arowBase = A + (size_t)(mBase + l16) * Kdim + half * 16;
  for (int k = 0; k < Kdim; k += 64) {
    union { v8i v; int2 d[4]; } af;
    const signed char* arow = arowBase + k;
    for (int j = 0; j < 4; j++) af.d[j] = *(const int2*)(arow + (j & 1) * 8 + (j >> 1) * 32);
    if (k + 128 < Kdim) __builtin_prefetch(arow + 128, 0, 1);
    for (int t = 0; t < 4; t++) {
      union { v8i v; int4 q[2]; } bf;
      const signed char* wrow = W + (size_t)(nBase + t * 16 + l16) * Kdim + k + half * 16;
      bf.q[0] = *(const int4*)(wrow);
      bf.q[1] = *(const int4*)(wrow + 32);
      acc[t] = __builtin_amdgcn_wmma_i32_16x16x64_iu8(true, af.v, true, bf.v,
                                                      acc[t], false, false);
    }
  }
  float ws = wscale[0];
  for (int t = 0; t < 4; t++) {
    int n = nBase + t * 16 + l16;
    float bn = bias[n];
    for (int v = 0; v < 8; v++) {
      int m = mBase + v + half * 8;
      float val = (float)acc[t][v] * (rowscale[m] * ws) + bn;
      if (MODE == 0) {
        float g = 0.5f * val * (1.f + erff(val * 0.70710678118f));
        outh[(size_t)m * Ndim + n] = (_Float16)g;
      } else {
        outf[(size_t)m * Ndim + n] = resid[(size_t)m * Ndim + n] + val;
      }
    }
  }
}

__global__ void k_build_vt(const _Float16* __restrict__ qkv, _Float16* __restrict__ vt) {
  size_t idx = (size_t)blockIdx.x * 256 + threadIdx.x;
  int s = idx & (SS - 1);
  int d = (idx >> 10) & (DH - 1);
  int h = (idx >> 16) & (HH - 1);
  int b = (int)(idx >> 20);
  vt[idx] = qkv[((size_t)(b * SS + s)) * (3 * DD) + 2 * DD + h * DH + d];
}

__global__ void k_attention(const _Float16* __restrict__ qkv,
                            const _Float16* __restrict__ vt,
                            _Float16* __restrict__ o) {
  __shared__ _Float16 lds_p[8][16 * 32];
  const int lane = threadIdx.x & 31;
  const int wave = threadIdx.x >> 5;
  const int half = lane >> 4;
  const int l16  = lane & 15;
  int tile = blockIdx.x * 8 + wave;
  int qt = tile & 63;
  int h  = (tile >> 6) & (HH - 1);
  int b  = tile >> 10;
  const size_t rowp = 3 * DD;

  F16Frag qa[2];
  {
    const _Float16* qrow = qkv + ((size_t)(b * SS) + qt * 16 + l16) * rowp + h * DH;
    for (int c = 0; c < 2; c++) {
      qa[c].s.lo = *(const v8h*)(qrow + c * 32 + half * 16);
      qa[c].s.hi = *(const v8h*)(qrow + c * 32 + half * 16 + 8);
    }
  }
  float mrow[8], lrow[8];
  v8f ofrag[4] = {};
  for (int v = 0; v < 8; v++) { mrow[v] = -1e30f; lrow[v] = 0.f; }
  const float sc = 0.125f;
  _Float16* myp = &lds_p[wave][0];

  for (int kb = 0; kb < SS; kb += 32) {
    v8f s0 = {}, s1 = {};
    for (int c = 0; c < 2; c++) {
      const _Float16* k0 = qkv + ((size_t)(b * SS) + kb + l16) * rowp + DD + h * DH + c * 32 + half * 16;
      const _Float16* k1 = qkv + ((size_t)(b * SS) + kb + 16 + l16) * rowp + DD + h * DH + c * 32 + half * 16;
      v16h kf0 = *(const v16h*)k0;
      v16h kf1 = *(const v16h*)k1;
      s0 = __builtin_amdgcn_wmma_f32_16x16x32_f16(false, qa[c].v, false, kf0, (short)0, s0, false, false);
      s1 = __builtin_amdgcn_wmma_f32_16x16x32_f16(false, qa[c].v, false, kf1, (short)0, s1, false, false);
    }
    float e0[8], e1[8];
    for (int v = 0; v < 8; v++) {
      float a0 = s0[v] * sc, a1 = s1[v] * sc;
      float mx = row16_max(fmaxf(a0, a1));
      float mnew = fmaxf(mrow[v], mx);
      float corr = __expf(mrow[v] - mnew);
      e0[v] = __expf(a0 - mnew);
      e1[v] = __expf(a1 - mnew);
      float ssum = row16_sum(e0[v] + e1[v]);
      lrow[v] = lrow[v] * corr + ssum;
      mrow[v] = mnew;
      for (int t = 0; t < 4; t++) ofrag[t][v] *= corr;
    }
    for (int v = 0; v < 8; v++) {
      int m = v + half * 8;
      myp[m * 32 + l16]      = (_Float16)e0[v];
      myp[m * 32 + 16 + l16] = (_Float16)e1[v];
    }
    asm volatile("s_wait_dscnt 0" ::: "memory");
    F16Frag pa;
    pa.s.lo = *(const v8h*)&myp[l16 * 32 + half * 16];
    pa.s.hi = *(const v8h*)&myp[l16 * 32 + half * 16 + 8];
    for (int t = 0; t < 4; t++) {
      const _Float16* vrow = vt + ((size_t)((b * HH + h) * DH + t * 16 + l16)) * SS + kb + half * 16;
      v16h vb = *(const v16h*)vrow;
      ofrag[t] = __builtin_amdgcn_wmma_f32_16x16x32_f16(false, pa.v, false, vb, (short)0, ofrag[t], false, false);
    }
  }
  for (int t = 0; t < 4; t++)
    for (int v = 0; v < 8; v++) {
      int m = qt * 16 + v + half * 8;
      int d = t * 16 + l16;
      o[((size_t)(b * SS) + m) * DD + h * DH + d] = (_Float16)(ofrag[t][v] / lrow[v]);
    }
}

extern "C" void kernel_launch(void* const* d_in, const int* in_sizes, int n_in,
                              void* d_out, int out_size, void* d_ws, size_t ws_size,
                              hipStream_t stream) {
  const float* x          = (const float*)d_in[0];
  const float* norm1_w    = (const float*)d_in[1];
  const float* in_proj_w  = (const float*)d_in[2];
  const float* in_proj_b  = (const float*)d_in[3];
  const float* out_proj_w = (const float*)d_in[4];
  const float* out_proj_b = (const float*)d_in[5];
  const float* norm2_w    = (const float*)d_in[6];
  const float* w1         = (const float*)d_in[7];
  const float* b1         = (const float*)d_in[8];
  const float* w2         = (const float*)d_in[9];
  const float* b2         = (const float*)d_in[10];
  float* out = (float*)d_out;

  char* base = (char*)d_ws;
  size_t off = 0;
  auto alloc = [&](size_t bytes) { void* p = base + off; off += (bytes + 255) & ~(size_t)255; return p; };
  _Float16*    f_xn   = (_Float16*)alloc((size_t)MM * DD * 2);
  _Float16*    f_wqkv = (_Float16*)alloc((size_t)3 * DD * DD * 2);
  _Float16*    f_wout = (_Float16*)alloc((size_t)DD * DD * 2);
  _Float16*    f_qkv  = (_Float16*)alloc((size_t)MM * 3 * DD * 2);
  _Float16*    f_vt   = (_Float16*)alloc((size_t)MM * DD * 2);
  _Float16*    f_o    = (_Float16*)alloc((size_t)MM * DD * 2);
  float*       f_x1   = (float*)alloc((size_t)MM * DD * 4);
  signed char* q_y    = (signed char*)alloc((size_t)MM * DD);
  float*       s_y    = (float*)alloc((size_t)MM * 4);
  signed char* q_w1   = (signed char*)alloc((size_t)FFN * DD);
  signed char* q_w2   = (signed char*)alloc((size_t)DD * FFN);
  float*       s_w    = (float*)alloc(4 * sizeof(float));
  _Float16*    f_h    = (_Float16*)alloc((size_t)MM * FFN * 2);
  signed char* q_h    = (signed char*)alloc((size_t)MM * FFN);
  float*       s_h    = (float*)alloc((size_t)MM * 4);

  hipMemsetAsync(s_w, 0, 2 * sizeof(float), stream);
  k_f32_to_f16<<<4096, 256, 0, stream>>>(in_proj_w, f_wqkv, 3 * DD * DD);
  k_f32_to_f16<<<4096, 256, 0, stream>>>(out_proj_w, f_wout, DD * DD);
  k_abs_sum<<<1024, 256, 0, stream>>>(w1, FFN * DD, s_w + 0);
  k_abs_sum<<<1024, 256, 0, stream>>>(w2, DD * FFN, s_w + 1);
  k_tern_quant<<<1024, 256, 0, stream>>>(w1, FFN * DD, s_w + 0, 1.f / (FFN * DD), q_w1, s_w + 2);
  k_tern_quant<<<1024, 256, 0, stream>>>(w2, DD * FFN, s_w + 1, 1.f / (DD * FFN), q_w2, s_w + 3);

  k_rmsnorm_f16<<<MM, 256, 0, stream>>>(x, norm1_w, f_xn);
  k_gemm_f16<0><<<dim3(3 * DD / 64, MM / 128), 256, 0, stream>>>(
      f_xn, f_wqkv, in_proj_b, nullptr, f_qkv, nullptr, 3 * DD, DD);
  k_build_vt<<<(MM * DD) / 256, 256, 0, stream>>>(f_qkv, f_vt);
  k_attention<<<(BB * HH * (SS / 16)) / 8, 256, 0, stream>>>(f_qkv, f_vt, f_o);
  k_gemm_f16<1><<<dim3(DD / 64, MM / 128), 256, 0, stream>>>(
      f_o, f_wout, out_proj_b, x, nullptr, f_x1, DD, DD);

  k_rmsnorm_q8<<<MM, 256, 0, stream>>>(f_x1, norm2_w, q_y, s_y);
  k_gemm_i8<0><<<dim3(FFN / 64, MM / 128), 256, 0, stream>>>(
      q_y, q_w1, s_y, s_w + 2, b1, nullptr, f_h, nullptr, FFN, DD);
  k_actquant_f16<<<MM, 256, 0, stream>>>(f_h, q_h, s_h, FFN);
  k_gemm_i8<1><<<dim3(DD / 64, MM / 128), 256, 0, stream>>>(
      q_h, q_w2, s_h, s_w + 3, b2, f_x1, nullptr, out, DD, FFN);
}
